// MMORE_GAT_11622181503326
// MI455X (gfx1250) — matrix operand order repaired and checked by reading, not compiled, not run on hardware
//
#include <hip/hip_runtime.h>
#include <cstdint>
#include <cstddef>

// ---------------------------------------------------------------------------
// MMORE/GAT forward for MI455X (gfx1250, wave32).
// GEMM stages on v_wmma_f32_16x16x32_f16; shared B panels double-buffered in
// LDS via GLOBAL_LOAD_ASYNC_TO_LDS_B128 (+ s_wait_asynccnt); epilogues use the
// CDNA5 C-frag layout (row = vgpr + 8*(lane/16), col = lane%16).
// ---------------------------------------------------------------------------

typedef _Float16 h16;
typedef __attribute__((ext_vector_type(16))) _Float16 v16h;
typedef __attribute__((ext_vector_type(8)))  float    v8f;
typedef int v4i_ __attribute__((vector_size(16)));

#define DEVINL __device__ __forceinline__

constexpr int Bb   = 64;
constexpr int Tt   = 20;
constexpr int NX   = 30;      // dx codes per visit
constexpr int NR   = 30;      // drug codes per visit
constexpr int DXV  = 10000;
constexpr int DRUGV= 3000;
constexpr int D    = 128;     // ONTOD == EHRD == ATTND == PTATTND
constexpr int BT   = Bb * Tt;         // 1280 rows
constexpr int VOC  = DXV + DRUGV;     // 13000
constexpr int VOCP = 13008;           // 813 * 16 (zero-padded rows in coWt)
constexpr int KPAD_DX   = 10016;      // 313 * 32
constexpr int KPAD_DRUG = 3008;       // 94 * 32
constexpr int DPL  = 500;

// ---- CDNA5 async global->LDS copy (16B per lane) ----
#if __has_builtin(__builtin_amdgcn_global_load_async_to_lds_b128)
#define ASYNC_COPY16(dst, src)                                               \
  __builtin_amdgcn_global_load_async_to_lds_b128(                            \
      (__attribute__((address_space(1))) v4i_*)(src),                        \
      (__attribute__((address_space(3))) v4i_*)(dst), 0, 0)
#else
#define ASYNC_COPY16(dst, src)                                               \
  asm volatile("global_load_async_to_lds_b128 %0, %1, off" ::                \
                   "v"((__attribute__((address_space(3))) void*)(dst)),      \
               "v"((__attribute__((address_space(1))) void*)(src))           \
               : "memory")
#endif

#if __has_builtin(__builtin_amdgcn_s_wait_asynccnt)
#define WAIT_ASYNC() __builtin_amdgcn_s_wait_asynccnt(0)
#else
#define WAIT_ASYNC() asm volatile("s_wait_asynccnt 0x0" ::: "memory")
#endif

DEVINL v8f wmma16(v16h a, v16h b, v8f c) {
  return __builtin_amdgcn_wmma_f32_16x16x32_f16(false, a, false, b,
                                                (short)0, c, false, false);
}

DEVINL float red16_sum(float v) {
  v += __shfl_xor(v, 1, 32);
  v += __shfl_xor(v, 2, 32);
  v += __shfl_xor(v, 4, 32);
  v += __shfl_xor(v, 8, 32);
  return v;
}

DEVINL float block_sum128(float v, float* red) {
#pragma unroll
  for (int m = 1; m < 32; m <<= 1) v += __shfl_xor(v, m, 32);
  int w = threadIdx.x >> 5;
  if ((threadIdx.x & 31) == 0) red[w] = v;
  __syncthreads();
  float s = red[0] + red[1] + red[2] + red[3];
  __syncthreads();
  return s;
}

DEVINL float block_max128(float v, float* red) {
#pragma unroll
  for (int m = 1; m < 32; m <<= 1) v = fmaxf(v, __shfl_xor(v, m, 32));
  int w = threadIdx.x >> 5;
  if ((threadIdx.x & 31) == 0) red[w] = v;
  __syncthreads();
  float s = fmaxf(fmaxf(red[0], red[1]), fmaxf(red[2], red[3]));
  __syncthreads();
  return s;
}

// ---------------------------------------------------------------------------
// init: zero pad rows of dxALL/drugALL
// ---------------------------------------------------------------------------
__global__ void k_init(float* dxALL, float* drugALL) {
  int t = threadIdx.x;
  if (t < D) {
    dxALL[(size_t)DXV * D + t]     = 0.f;
    drugALL[(size_t)DRUGV * D + t] = 0.f;
  }
}

// ---------------------------------------------------------------------------
// transpose + f32->f16 convert:  in (K,Nin) row-major -> out (Nout, Kpad) f16
// zero-pads k in [K,Kpad) and n in [Nin,Nout).
// ---------------------------------------------------------------------------
__global__ __launch_bounds__(256) void k_tcvt(const float* __restrict__ in,
                                              h16* __restrict__ out,
                                              int K, int Nin, int Nout,
                                              int Kpad) {
  size_t total = (size_t)Nout * Kpad;
  for (size_t e = (size_t)blockIdx.x * blockDim.x + threadIdx.x; e < total;
       e += (size_t)gridDim.x * blockDim.x) {
    int n = (int)(e / Kpad);
    int k = (int)(e % Kpad);
    out[e] = (h16)((k < K && n < Nin) ? in[(size_t)k * Nin + n] : 0.f);
  }
}

// ---------------------------------------------------------------------------
// GRAM ontology attention. One wave handles 2 leaves (16 rows = 2x8 ancestor
// pairs); M=16, N=128 (8 tiles), K=256 (8 steps). aWt is pre-transposed f16.
// ---------------------------------------------------------------------------
__global__ __launch_bounds__(256) void k_onto(const float* __restrict__ emb,
                                              const int* __restrict__ anc,
                                              const h16* __restrict__ aWt,
                                              const float* __restrict__ ab,
                                              const float* __restrict__ cW,
                                              float* __restrict__ outALL,
                                              int V) {
  const int gwave = (int)((blockIdx.x * blockDim.x + threadIdx.x) >> 5);
  const int lane  = threadIdx.x & 31;
  const int leaf0 = gwave * 2;
  if (leaf0 >= V) return;
  const int lm = lane & 15, half = lane >> 4, off = half * 8;
  const int leaf  = leaf0 + (lm >> 3);   // A-row lm -> leaf, ancestor
  const int aidx  = lm & 7;
  const int ancid = anc[leaf * 8 + aidx];
  const float* lrow = emb + (size_t)leaf * D;
  const float* arow = emb + (size_t)ancid * D;

  v8f acc[8] = {};
  for (int ks = 0; ks < 8; ++ks) {
    const int kb = ks * 32;
    v16h af;
#pragma unroll
    for (int i = 0; i < 8; ++i) {
      int k0 = kb + half * 16 + i;
      int k1 = kb + half * 16 + 8 + i;
      af[i]     = (h16)((k0 < D) ? lrow[k0] : arow[k0 - D]);
      af[i + 8] = (h16)((k1 < D) ? lrow[k1] : arow[k1 - D]);
    }
#pragma unroll
    for (int nt = 0; nt < 8; ++nt) {
      const h16* bp = aWt + (size_t)(nt * 16 + lm) * 256 + kb + half * 16;
      acc[nt] = wmma16(af, *(const v16h*)bp, acc[nt]);
    }
  }
  // s[m] = tanh(h)·cW  (cb is softmax-invariant)
  float sc[8] = {};
#pragma unroll
  for (int nt = 0; nt < 8; ++nt) {
    int c = nt * 16 + lm;
    float bias = ab[c], w = cW[c];
#pragma unroll
    for (int r = 0; r < 8; ++r) sc[r] += tanhf(acc[nt][r] + bias) * w;
  }
#pragma unroll
  for (int r = 0; r < 8; ++r) sc[r] = red16_sum(sc[r]);
  // per-lane softmax over this lane's leaf (rows r + 8*half)
  float mx = sc[0];
#pragma unroll
  for (int r = 1; r < 8; ++r) mx = fmaxf(mx, sc[r]);
  float w8[8], S = 0.f;
#pragma unroll
  for (int r = 0; r < 8; ++r) { w8[r] = __expf(sc[r] - mx); S += w8[r]; }
  float invS = 1.f / S;
  const int myleaf = leaf0 + half;
#pragma unroll
  for (int j = 0; j < 8; ++j) {
    int d = lm + 16 * j;
    float o = 0.f;
#pragma unroll
    for (int a = 0; a < 8; ++a)
      o += (w8[a] * invS) * emb[(size_t)anc[myleaf * 8 + a] * D + d];
    outALL[(size_t)myleaf * D + d] = o;
  }
}

// ---------------------------------------------------------------------------
// EHR visit embedding: EHRV[bt] = l2n(sum dxE) + l2n(sum drugE)
// ---------------------------------------------------------------------------
__global__ __launch_bounds__(128) void k_ehrv(const int* __restrict__ dxseq,
                                              const int* __restrict__ drugseq,
                                              const float* __restrict__ edx,
                                              const float* __restrict__ edrug,
                                              float* __restrict__ EHRV) {
  __shared__ float red[4];
  const int bt = blockIdx.x, d = threadIdx.x;
  float a1 = 0.f, a2 = 0.f;
  for (int n = 0; n < NX; ++n)
    a1 += edx[(size_t)dxseq[bt * NX + n] * D + d];
  for (int n = 0; n < NR; ++n)
    a2 += edrug[(size_t)drugseq[bt * NR + n] * D + d];
  float s1 = block_sum128(a1 * a1, red);
  float s2 = block_sum128(a2 * a2, red);
  float v = a1 / fmaxf(sqrtf(s1), 1e-12f) + a2 / fmaxf(sqrtf(s2), 1e-12f);
  EHRV[(size_t)bt * D + d] = v;
}

// ---------------------------------------------------------------------------
// onehot (T,B,V) x ALL (V,128) -> (B*T,128).  Bt is (128, Kpad) f16.
// One wave = one 16-row tile x all 8 N-tiles.  The shared (per-block) 32xK
// B panel is double-buffered in LDS via async global->LDS copies.
// ---------------------------------------------------------------------------
__global__ __launch_bounds__(256) void k_ontoV(const float* __restrict__ onehot,
                                               const h16* __restrict__ Bt,
                                               float* __restrict__ outV,
                                               int V, int Ktiles, int Kpad) {
  __shared__ h16 Bst[2][128 * 32];          // 2 x 8 KB, rows = N cols of Bt
  const int tid  = threadIdx.x;
  const int wave = blockIdx.x * 8 + (tid >> 5);
  const int lane = tid & 31;
  const int lm = lane & 15, half = lane >> 4, off = half * 8;
  const int r = wave * 16 + lm;             // A-row index (b*T + t)
  const int b = r / Tt, t = r % Tt;
  const float* arow = onehot + ((size_t)t * Bb + b) * V;

  // async stage of one 128(N) x 32(K) panel: 512 x 16B chunks, 2 per thread
  auto stage = [&](int kt, int buf) {
    const int kb = kt * 32;
    const int n = tid >> 1;
    const int ch = (tid & 1) * 2;
#pragma unroll
    for (int q = 0; q < 2; ++q) {
      const h16* src = Bt + (size_t)n * Kpad + kb + (ch + q) * 8;
      ASYNC_COPY16(&Bst[buf][n * 32 + (ch + q) * 8], src);
    }
  };

  stage(0, 0);
  WAIT_ASYNC();
  __syncthreads();

  v8f acc[8] = {};
  for (int kt = 0; kt < Ktiles; ++kt) {
    const int buf = kt & 1;
    if (kt + 1 < Ktiles) stage(kt + 1, buf ^ 1);
    const int kb = kt * 32;
    __builtin_prefetch(arow + kb + 64, 0, 1);
    v16h af;
#pragma unroll
    for (int i = 0; i < 8; ++i) {
      int k0 = kb + half * 16 + i;
      int k1 = kb + half * 16 + 8 + i;
      af[i]     = (h16)((k0 < V) ? arow[k0] : 0.f);
      af[i + 8] = (h16)((k1 < V) ? arow[k1] : 0.f);
    }
#pragma unroll
    for (int nt = 0; nt < 8; ++nt) {
      const h16* bp = &Bst[buf][(nt * 16 + lm) * 32 + half * 16];
      acc[nt] = wmma16(af, *(const v16h*)bp, acc[nt]);
    }
    WAIT_ASYNC();       // next panel resident
    __syncthreads();    // everyone done reading current panel
  }
#pragma unroll
  for (int nt = 0; nt < 8; ++nt) {
    int c = nt * 16 + lm;
#pragma unroll
    for (int rr = 0; rr < 8; ++rr) {
      int row = wave * 16 + rr + 8 * half;
      outV[(size_t)row * D + c] = acc[nt][rr];
    }
  }
}

// ---------------------------------------------------------------------------
// vs = [ l2n(dxontoV) + l2n(drugontoV) | EHRV ]
// ---------------------------------------------------------------------------
__global__ __launch_bounds__(128) void k_vs(const float* __restrict__ dxoV,
                                            const float* __restrict__ droV,
                                            const float* __restrict__ EHRV,
                                            float* __restrict__ vs) {
  __shared__ float red[4];
  const int bt = blockIdx.x, d = threadIdx.x;
  float x1 = dxoV[(size_t)bt * D + d];
  float x2 = droV[(size_t)bt * D + d];
  float s1 = block_sum128(x1 * x1, red);
  float s2 = block_sum128(x2 * x2, red);
  vs[(size_t)bt * 256 + d] =
      x1 / fmaxf(sqrtf(s1), 1e-12f) + x2 / fmaxf(sqrtf(s2), 1e-12f);
  vs[(size_t)bt * 256 + D + d] = EHRV[(size_t)bt * D + d];
}

// ---------------------------------------------------------------------------
// co-occur: logits = EHRV @ co_W + co_b over 13000 cols; streaming softmax
// loss per row.  One wave = 16 rows; coWt is (13008,128) f16, zero padded.
// The shared 16x128 B panel per tile is async double-buffered in LDS.
// ---------------------------------------------------------------------------
__global__ __launch_bounds__(256) void k_cooccur(const float* __restrict__ EHRV,
                                                 const h16* __restrict__ coWt,
                                                 const float* __restrict__ cob,
                                                 const float* __restrict__ dxoh,
                                                 const float* __restrict__ drugoh,
                                                 float* __restrict__ rowloss) {
  __shared__ h16 Bs[2][16 * 128];           // 2 x 4 KB
  const int tid  = threadIdx.x;
  const int wave = blockIdx.x * 8 + (tid >> 5);
  const int lane = tid & 31;
  const int lm = lane & 15, half = lane >> 4, off = half * 8;
  const float* arow = EHRV + (size_t)(wave * 16 + lm) * D;

  v16h afr[4];
#pragma unroll
  for (int ks = 0; ks < 4; ++ks) {
    const int kb = ks * 32;
#pragma unroll
    for (int i = 0; i < 8; ++i) {
      afr[ks][i]     = (h16)arow[kb + half * 16 + i];
      afr[ks][i + 8] = (h16)arow[kb + half * 16 + 8 + i];
    }
  }

  auto stage = [&](int tile, int buf) {     // 16 cols x 128 halves = 4 KB
    const h16* src = coWt + (size_t)tile * 16 * D + tid * 8;
    ASYNC_COPY16(&Bs[buf][tid * 8], src);
  };

  const int ntiles = VOCP / 16;             // 813
  stage(0, 0);
  WAIT_ASYNC();
  __syncthreads();

  float sume[8] = {}, dot[8] = {}, osum[8] = {};
  for (int tile = 0; tile < ntiles; ++tile) {
    const int buf = tile & 1;
    if (tile + 1 < ntiles) stage(tile + 1, buf ^ 1);
    const int c = tile * 16 + lm;
    const bool valid = (c < VOC);
    v8f C = {};
#pragma unroll
    for (int ks = 0; ks < 4; ++ks)
      C = wmma16(afr[ks], *(const v16h*)&Bs[buf][lm * D + ks * 32 + half * 16],
                 C);
    const float cb = valid ? cob[c] : 0.f;
#pragma unroll
    for (int rr = 0; rr < 8; ++rr) {
      float x = C[rr] + cb;
      float e = valid ? __expf(x) : 0.f;
      sume[rr] += e;
      if (valid) {
        int row = wave * 16 + rr + 8 * half;
        int b = row / Tt, t = row % Tt;
        float o = (c < DXV)
                      ? dxoh[((size_t)t * Bb + b) * DXV + c]
                      : drugoh[((size_t)t * Bb + b) * DRUGV + (c - DXV)];
        dot[rr]  += o * x;
        osum[rr] += o;
      }
    }
    WAIT_ASYNC();
    __syncthreads();
  }
#pragma unroll
  for (int rr = 0; rr < 8; ++rr) {
    float S = red16_sum(sume[rr]);
    float Dt = red16_sum(dot[rr]);
    float O = red16_sum(osum[rr]);
    if (lm == 0) {
      int row = wave * 16 + rr + 8 * half;
      rowloss[row] = O * __logf(S) - Dt;    // -sum(onehot * log softmax)
    }
  }
}

__global__ __launch_bounds__(256) void k_loss(const float* __restrict__ rowloss,
                                              float* __restrict__ lossOut) {
  __shared__ float red[8];
  float v = 0.f;
  for (int i = threadIdx.x; i < BT; i += 256) v += rowloss[i];
#pragma unroll
  for (int m = 1; m < 32; m <<= 1) v += __shfl_xor(v, m, 32);
  int w = threadIdx.x >> 5;
  if ((threadIdx.x & 31) == 0) red[w] = v;
  __syncthreads();
  if (threadIdx.x == 0) {
    float s = 0.f;
    for (int i = 0; i < 8; ++i) s += red[i];
    lossOut[0] = s * (10.f / Bb);
  }
}

// ---------------------------------------------------------------------------
// Patient-level attention. One block per (b,t): stage the 64x512 f16 A tile
// (rows = 60 codes + 4 zero pad, cols = [vs(256)|EHR(128)|onto(128)]) in LDS,
// 4 waves x (16 rows x 128 cols, K=512) WMMA, then softmax over 60 and vs2.
// ---------------------------------------------------------------------------
__global__ __launch_bounds__(128) void k_pattn(const int* __restrict__ dxseq,
                                               const int* __restrict__ drugseq,
                                               const float* __restrict__ vs,
                                               const float* __restrict__ edx,
                                               const float* __restrict__ edrug,
                                               const float* __restrict__ dxALL,
                                               const float* __restrict__ drugALL,
                                               const h16* __restrict__ attnWt,
                                               const float* __restrict__ attnb,
                                               const float* __restrict__ combW,
                                               const float* __restrict__ combb,
                                               float* __restrict__ vs2) {
  __shared__ h16 Atile[64 * 512];      // 64 KB
  const int bt = blockIdx.x;
  // ---- stage A tile (f32 -> f16) ----
  for (int e = threadIdx.x; e < 64 * 512; e += 128) {
    const int n = e >> 9, k = e & 511;
    float val = 0.f;
    if (n < 60) {
      if (k < 256) {
        val = vs[(size_t)bt * 256 + k];
      } else if (k < 384) {
        int kk = k - 256;
        val = (n < NX) ? edx[(size_t)dxseq[bt * NX + n] * D + kk]
                       : edrug[(size_t)drugseq[bt * NR + (n - NX)] * D + kk];
      } else {
        int kk = k - 384;
        val = (n < NX) ? dxALL[(size_t)dxseq[bt * NX + n] * D + kk]
                       : drugALL[(size_t)drugseq[bt * NR + (n - NX)] * D + kk];
      }
    }
    Atile[e] = (h16)val;
  }
  __syncthreads();

  const int wv = threadIdx.x >> 5, lane = threadIdx.x & 31;
  const int lm = lane & 15, half = lane >> 4, off = half * 8;
  const h16* arow = &Atile[(wv * 16 + lm) * 512];

  v8f acc[8] = {};
  for (int ks = 0; ks < 16; ++ks) {
    const int kb = ks * 32;
    v16h af;
#pragma unroll
    for (int i = 0; i < 8; ++i) {
      af[i]     = arow[kb + half * 16 + i];
      af[i + 8] = arow[kb + half * 16 + 8 + i];
    }
#pragma unroll
    for (int nt = 0; nt < 8; ++nt) {
      const h16* bp = attnWt + (size_t)(nt * 16 + lm) * 512 + kb + half * 16;
      acc[nt] = wmma16(af, *(const v16h*)bp, acc[nt]);
    }
  }
  // pre[m] = tanh(h)·comb_W
  float pr[8] = {};
#pragma unroll
  for (int nt = 0; nt < 8; ++nt) {
    int c = nt * 16 + lm;
    float bias = attnb[c], w = combW[c];
#pragma unroll
    for (int rr = 0; rr < 8; ++rr) pr[rr] += tanhf(acc[nt][rr] + bias) * w;
  }
#pragma unroll
  for (int rr = 0; rr < 8; ++rr) pr[rr] = red16_sum(pr[rr]);

  __syncthreads();                      // all waves done reading Atile frags
  float* preS = (float*)&Atile[63 * 512];   // reuse pad row as scratch
  if (lm == 0) {
#pragma unroll
    for (int rr = 0; rr < 8; ++rr)
      preS[wv * 16 + rr + 8 * half] = pr[rr] + combb[0];
  }
  __syncthreads();

  // softmax over 60 codes; vs2[d] = sum_n attn[n] * dxdrug[n][d]
  const int d = threadIdx.x;            // dims d (EHR) and d+128 (onto)
  float mx = -1e30f;
  for (int n = 0; n < 60; ++n) mx = fmaxf(mx, preS[n]);
  float S = 0.f;
  for (int n = 0; n < 60; ++n) S += __expf(preS[n] - mx);
  float o1 = 0.f, o2 = 0.f;
  for (int n = 0; n < 60; ++n) {
    float a = __expf(preS[n] - mx) / S;
    o1 += a * (float)Atile[n * 512 + 256 + d];
    o2 += a * (float)Atile[n * 512 + 384 + d];
  }
  vs2[(size_t)bt * 256 + d]       = o1;
  vs2[(size_t)bt * 256 + 128 + d] = o2;
}

// ---------------------------------------------------------------------------
// heads: vs_dp = l2n(vs2); DP = softmax(tanh(vs_dp)@dp_W + dp_b);
// read = sigmoid(vs_dp @ read_W + read_b)
// ---------------------------------------------------------------------------
__global__ __launch_bounds__(128) void k_heads(const float* __restrict__ vs2,
                                               const float* __restrict__ dpW,
                                               const float* __restrict__ dpb,
                                               const float* __restrict__ readW,
                                               const float* __restrict__ readb,
                                               float* __restrict__ out) {
  __shared__ float vsd[256], th[256], lg[512], red[4];
  const int bt = blockIdx.x, tid = threadIdx.x;
  float x0 = vs2[(size_t)bt * 256 + tid];
  float x1 = vs2[(size_t)bt * 256 + 128 + tid];
  float ss = block_sum128(x0 * x0 + x1 * x1, red);
  float inv = 1.f / fmaxf(sqrtf(ss), 1e-12f);
  vsd[tid] = x0 * inv;      vsd[tid + 128] = x1 * inv;
  th[tid] = tanhf(vsd[tid]); th[tid + 128] = tanhf(vsd[tid + 128]);
  __syncthreads();
  for (int o = tid; o < DPL; o += 128) {
    float a = dpb[o];
    for (int k = 0; k < 256; ++k) a += th[k] * dpW[(size_t)k * DPL + o];
    lg[o] = a;
  }
  __syncthreads();
  float lmax = -1e30f;
  for (int o = tid; o < DPL; o += 128) lmax = fmaxf(lmax, lg[o]);
  float bm = block_max128(lmax, red);
  float ls = 0.f;
  for (int o = tid; o < DPL; o += 128) ls += __expf(lg[o] - bm);
  float bs = block_sum128(ls, red);
  for (int o = tid; o < DPL; o += 128)
    out[BT + (size_t)bt * DPL + o] = __expf(lg[o] - bm) / bs;
  float rp = vsd[tid] * readW[tid] + vsd[tid + 128] * readW[tid + 128];
  float rs = block_sum128(rp, red);
  if (tid == 0) out[bt] = 1.f / (1.f + __expf(-(rs + readb[0])));
}

// ---------------------------------------------------------------------------
extern "C" void kernel_launch(void* const* d_in, const int* in_sizes, int n_in,
                              void* d_out, int out_size, void* d_ws,
                              size_t ws_size, hipStream_t stream) {
  (void)in_sizes; (void)n_in; (void)out_size; (void)ws_size;
  const int*   dxseqs   = (const int*)d_in[0];
  const int*   drugseqs = (const int*)d_in[1];
  const float* dxoh     = (const float*)d_in[2];
  const float* drugoh   = (const float*)d_in[3];
  const int*   dxanc    = (const int*)d_in[5];
  const int*   druganc  = (const int*)d_in[7];
  const float* dxemb    = (const float*)d_in[8];
  const float* dxaW     = (const float*)d_in[9];
  const float* dxab     = (const float*)d_in[10];
  const float* dxcW     = (const float*)d_in[11];
  const float* drugemb  = (const float*)d_in[13];
  const float* drugaW   = (const float*)d_in[14];
  const float* drugab   = (const float*)d_in[15];
  const float* drugcW   = (const float*)d_in[16];
  const float* edx      = (const float*)d_in[18];
  const float* edrug    = (const float*)d_in[19];
  const float* coW      = (const float*)d_in[20];
  const float* cob      = (const float*)d_in[21];
  const float* attnW    = (const float*)d_in[22];
  const float* attnb    = (const float*)d_in[23];
  const float* combW    = (const float*)d_in[24];
  const float* combb    = (const float*)d_in[25];
  const float* dpW      = (const float*)d_in[26];
  const float* dpb      = (const float*)d_in[27];
  const float* readW    = (const float*)d_in[28];
  const float* readb    = (const float*)d_in[29];
  float* out = (float*)d_out;

  char* ws = (char*)d_ws;
  size_t cur = 0;
  auto take = [&](size_t bytes) -> char* {
    char* p = ws + cur;
    cur = (cur + bytes + 255) & ~(size_t)255;
    return p;
  };
  float* dxALL     = (float*)take((size_t)(DXV + 1) * D * 4);
  float* drugALL   = (float*)take((size_t)(DRUGV + 1) * D * 4);
  h16*   aWdxt     = (h16*)take((size_t)D * 256 * 2);
  h16*   aWdrt     = (h16*)take((size_t)D * 256 * 2);
  h16*   attnWt    = (h16*)take((size_t)D * 512 * 2);
  h16*   coWt      = (h16*)take((size_t)VOCP * D * 2);
  h16*   dxALLt    = (h16*)take((size_t)D * KPAD_DX * 2);
  h16*   drugALLt  = (h16*)take((size_t)D * KPAD_DRUG * 2);
  float* EHRV      = (float*)take((size_t)BT * D * 4);
  float* dxoV      = (float*)take((size_t)BT * D * 4);
  float* droV      = (float*)take((size_t)BT * D * 4);
  float* vs        = (float*)take((size_t)BT * 256 * 4);
  float* vs2       = (float*)take((size_t)BT * 256 * 4);
  float* rowloss   = (float*)take((size_t)BT * 4);
  float* lossOut   = out + BT + (size_t)BT * DPL;

  k_init<<<1, 128, 0, stream>>>(dxALL, drugALL);

  // weight transposes (f32 -> f16, lane-per-column layout)
  k_tcvt<<<128, 256, 0, stream>>>(dxaW, aWdxt, 256, D, D, 256);
  k_tcvt<<<128, 256, 0, stream>>>(drugaW, aWdrt, 256, D, D, 256);
  k_tcvt<<<256, 256, 0, stream>>>(attnW, attnWt, 512, D, D, 512);
  k_tcvt<<<2048, 256, 0, stream>>>(coW, coWt, D, VOC, VOCP, D);

  // GRAM ontology attention (WMMA)
  k_onto<<<625, 256, 0, stream>>>(dxemb, dxanc, aWdxt, dxab, dxcW, dxALL, DXV);
  k_onto<<<188, 256, 0, stream>>>(drugemb, druganc, aWdrt, drugab, drugcW,
                                  drugALL, DRUGV);

  // transpose runtime-produced ALL tables for the big einsum
  k_tcvt<<<2048, 256, 0, stream>>>(dxALL, dxALLt, DXV, D, D, KPAD_DX);
  k_tcvt<<<1024, 256, 0, stream>>>(drugALL, drugALLt, DRUGV, D, D, KPAD_DRUG);

  k_ehrv<<<BT, 128, 0, stream>>>(dxseqs, drugseqs, edx, edrug, EHRV);

  // onehot @ ALL (WMMA + async LDS double buffering)
  k_ontoV<<<10, 256, 0, stream>>>(dxoh, dxALLt, dxoV, DXV, KPAD_DX / 32,
                                  KPAD_DX);
  k_ontoV<<<10, 256, 0, stream>>>(drugoh, drugALLt, droV, DRUGV,
                                  KPAD_DRUG / 32, KPAD_DRUG);

  k_vs<<<BT, 128, 0, stream>>>(dxoV, droV, EHRV, vs);

  // co-occur softmax loss (WMMA + async LDS double buffering)
  k_cooccur<<<10, 256, 0, stream>>>(EHRV, coWt, cob, dxoh, drugoh, rowloss);
  k_loss<<<1, 256, 0, stream>>>(rowloss, lossOut);

  // patient attention (WMMA, K=512)
  k_pattn<<<BT, 128, 0, stream>>>(dxseqs, drugseqs, vs, edx, edrug, dxALL,
                                  drugALL, attnWt, attnb, combW, combb, vs2);

  // heads
  k_heads<<<BT, 128, 0, stream>>>(vs2, dpW, dpb, readW, readb, out);
}
